// Model_74397423501811
// MI455X (gfx1250) — matrix operand order repaired and checked by reading, not compiled, not run on hardware
//
#include <hip/hip_runtime.h>
#include <hip/hip_bf16.h>
#include <math.h>

// ---------------- problem constants ----------------
#define BATCH   128
#define TSEQ    512
#define DMODEL  512
#define PREDN   96
#define DECT    144          // LABEL+PRED
#define DECROW  47           // -(PRED+1) row of x_dec
#define NWG     32           // grid: workgroups (one per WGP, 264KB LDS each)
#define NSL     64           // N-slice per WG = 16 hidden units x 4 gates
#define NTPB    256          // 8 wave32 per WG

// ---------------- workspace layout (bytes) ----------------
#define WS_BAR      0
#define WS_FB       1024                         // 128*7 f32 feedback
#define WS_BIAS     8192                         // 4*2048 f32 packed biases
#define WS_DWIH0    49152                        // 2048*7 f32 packed dec Wih0
#define WS_H        131072                       // 4 x (128*512 bf16) ping-pong h
#define HBUF_ELEMS  (BATCH*DMODEL)
#define MAT_ELEMS   (2048*512)
#define WS_PACK     (WS_H + (size_t)4*HBUF_ELEMS*2)
#define WS_EMB      (WS_PACK + (size_t)7*MAT_ELEMS*2)   // 128*512*512 bf16 = 64MB

// ---------------- LDS layout (bytes) ----------------
#define SM_W        262144   // 4 weight slices [64][512] bf16, pinned whole run
#define SM_XTRA     8192     // dec Wih0 slice (448 f32) + fb stage (896 f32)
#define SMEM_TOTAL  (SM_W + SM_XTRA)

typedef __attribute__((ext_vector_type(16))) __bf16 v16bf;
typedef __attribute__((ext_vector_type(8)))  __bf16 v8bf;
typedef __attribute__((ext_vector_type(8)))  float  v8f;

union AFrag { v16bf v; v8bf h[2]; };

__device__ __forceinline__ float sigf(float x) { return 1.0f / (1.0f + __expf(-x)); }

// ---------------- split grid barrier ----------------
// gsig: release fence (flush stores to L2) + arrival atomic.
// gwait: spin on agent-scope atomic (bypasses non-coherent WGP$), then
// acquire fence (global_inv) so post-wait h loads can't hit stale WGP$ lines.
// Cumulative counter: no WG can run more than one barrier ahead, so
// independent GEMMs placed between gsig and gwait overlap the wait window.
__device__ __forceinline__ void gsig(unsigned* bar) {
  __threadfence();
  __syncthreads();
  if (threadIdx.x == 0)
    __hip_atomic_fetch_add(bar, 1u, __ATOMIC_RELEASE, __HIP_MEMORY_SCOPE_AGENT);
}
__device__ __forceinline__ void gwait(unsigned* bar, unsigned& phase) {
  if (threadIdx.x == 0) {
    ++phase;
    const unsigned target = phase * (unsigned)NWG;
    while (__hip_atomic_load(bar, __ATOMIC_RELAXED, __HIP_MEMORY_SCOPE_AGENT) < target) {
      __builtin_amdgcn_s_sleep(1);
    }
  }
  __syncthreads();
  __threadfence();
}
__device__ __forceinline__ void gbar(unsigned* bar, unsigned& phase) {
  gsig(bar);
  gwait(bar, phase);
}

// ---------------- bulk copy of a 64KB weight slice into LDS ----------------
__device__ __forceinline__ void load_wlds(__bf16* dst, const __bf16* src, int wg, int tid) {
  const unsigned* s = (const unsigned*)(src + (size_t)wg * NSL * DMODEL);
  unsigned* d = (unsigned*)dst;
  for (int i = tid; i < NSL * DMODEL / 2; i += NTPB) d[i] = s[i];
}

// ---------------- bf16 WMMA GEMM: acc[4] += A[128x512] * Wslice[512x64] ----------------
// A fragments loaded directly from global (row-major, stride ars); B fragments
// from LDS-pinned weights.  All loads for a K-step issued before its 4 WMMAs.
__device__ __forceinline__ void gemm_bf16(const __bf16* __restrict__ Ag, size_t ars,
                                          const __bf16* __restrict__ Wlds,
                                          v8f* acc, int wave, int lq, int half) {
  const __bf16* __restrict__ arow = Ag + (size_t)(wave * 16 + lq) * ars + half * 16;
  const __bf16* __restrict__ wrow = Wlds + (size_t)lq * DMODEL + half * 16;
#pragma unroll 4
  for (int kb = 0; kb < 16; ++kb) {   // 16 K-steps of 32
    AFrag a;
    a.h[0] = *(const v8bf*)(arow + kb * 32);
    a.h[1] = *(const v8bf*)(arow + kb * 32 + 8);
    v16bf b0 = *(const v16bf*)(wrow + 0 * 16 * DMODEL + kb * 32);
    v16bf b1 = *(const v16bf*)(wrow + 1 * 16 * DMODEL + kb * 32);
    v16bf b2 = *(const v16bf*)(wrow + 2 * 16 * DMODEL + kb * 32);
    v16bf b3 = *(const v16bf*)(wrow + 3 * 16 * DMODEL + kb * 32);
    acc[0] = __builtin_amdgcn_wmma_f32_16x16x32_bf16(false, a.v, false, b0, (short)0, acc[0], false, false);
    acc[1] = __builtin_amdgcn_wmma_f32_16x16x32_bf16(false, a.v, false, b1, (short)0, acc[1], false, false);
    acc[2] = __builtin_amdgcn_wmma_f32_16x16x32_bf16(false, a.v, false, b2, (short)0, acc[2], false, false);
    acc[3] = __builtin_amdgcn_wmma_f32_16x16x32_bf16(false, a.v, false, b3, (short)0, acc[3], false, false);
  }
}

// ---------------- fused LSTM gate nonlinearity + cell update + h store ----------------
__device__ __forceinline__ void lstm_update(v8f* acc, float* c, __bf16* hw,
                                            int wg, int wave, int lq, int half) {
#pragma unroll
  for (int v = 0; v < 8; ++v) {
    const float ig = sigf(acc[0][v]);
    const float fg = sigf(acc[1][v]);
    const float gg = tanhf(acc[2][v]);
    const float og = sigf(acc[3][v]);
    const float cn = fg * c[v] + ig * gg;
    c[v] = cn;
    const float hn = og * tanhf(cn);
    const int b = wave * 16 + v + 8 * half;
    hw[(size_t)b * DMODEL + wg * 16 + lq] = (__bf16)hn;
  }
}

__device__ __forceinline__ void acc_init(v8f* acc, const float* br) {
#pragma unroll
  for (int nt = 0; nt < 4; ++nt)
#pragma unroll
    for (int v = 0; v < 8; ++v) acc[nt][v] = br[nt];
}
__device__ __forceinline__ void acc_copy(v8f* dst, const v8f* src) {
#pragma unroll
  for (int nt = 0; nt < 4; ++nt) dst[nt] = src[nt];
}

// ---------------- projection: out = h1 @ proj_w + proj_b, also refresh feedback ----------------
__device__ __forceinline__ void do_proj(const __bf16* h1w, const float* proj_w,
                                        const float* proj_b, float* out, float* fb,
                                        int td, int wg, int tid) {
  const int wave = tid >> 5, lane = tid & 31;
  for (int oi = wave; oi < 28; oi += 8) {
    const int b = wg * 4 + oi / 7, k = oi % 7;
    float s = 0.f;
    for (int d = lane; d < DMODEL; d += 32)
      s += (float)h1w[(size_t)b * DMODEL + d] * proj_w[d * 7 + k];
#pragma unroll
    for (int off = 16; off > 0; off >>= 1) s += __shfl_down(s, off, 32);
    if (lane == 0) {
      const float r = s + proj_b[k];
      out[(size_t)b * (PREDN * 7) + td * 7 + k] = r;
      fb[b * 7 + k] = r;
    }
  }
}

// ================= prep kernel 1: embedding (circular conv + time emb) -> bf16 =================
__global__ void embed_kernel(const float* __restrict__ x_enc, const float* __restrict__ x_mark,
                             const float* __restrict__ conv_w, const float* __restrict__ time_w,
                             __bf16* __restrict__ emb) {
  const int bt = blockIdx.x, b = bt / TSEQ, t = bt % TSEQ;
  float xv[3][7], mv[4];
#pragma unroll
  for (int j = 0; j < 3; ++j) {
    int tt = t + j - 1;
    if (tt < 0) tt += TSEQ;
    if (tt >= TSEQ) tt -= TSEQ;
#pragma unroll
    for (int c = 0; c < 7; ++c) xv[j][c] = x_enc[((size_t)b * TSEQ + tt) * 7 + c];
  }
#pragma unroll
  for (int f = 0; f < 4; ++f) mv[f] = x_mark[((size_t)b * TSEQ + t) * 4 + f];
  for (int d = threadIdx.x; d < DMODEL; d += NTPB) {
    float s = 0.f;
#pragma unroll
    for (int j = 0; j < 3; ++j)
#pragma unroll
      for (int c = 0; c < 7; ++c) s += xv[j][c] * conv_w[(j * 7 + c) * DMODEL + d];
#pragma unroll
    for (int f = 0; f < 4; ++f) s += mv[f] * time_w[f * DMODEL + d];
    emb[((size_t)b * TSEQ + t) * DMODEL + d] = (__bf16)s;
  }
}

// ================= prep kernel 2: permute+convert one weight matrix to packed bf16 =================
// packed[(wg*64 + gate*16 + jj)][k] = W[gate*512 + wg*16 + jj][k]
__global__ void pack_mat(const float* __restrict__ src, __bf16* __restrict__ dst) {
  const int drow = blockIdx.x;                 // 0..2047
  const int wg = drow >> 6, nl = drow & 63, gate = nl >> 4, jj = nl & 15;
  const int srow = gate * 512 + wg * 16 + jj;
  for (int k = threadIdx.x; k < DMODEL; k += NTPB)
    dst[(size_t)drow * DMODEL + k] = (__bf16)src[(size_t)srow * DMODEL + k];
}

// ================= prep kernel 3: packed biases (bih+bhh) + dec Wih0 slice =================
__global__ void pack_misc(const float* e0i, const float* e0h, const float* e1i, const float* e1h,
                          const float* d0i, const float* d0h, const float* d1i, const float* d1h,
                          const float* dwih0, float* biasp, float* dwih0p) {
  const int i = blockIdx.x * blockDim.x + threadIdx.x;
  if (i < 4 * 2048) {
    const int layer = i >> 11, d = i & 2047;
    const int wg = d >> 6, nl = d & 63, gate = nl >> 4, jj = nl & 15;
    const int s = gate * 512 + wg * 16 + jj;
    float v;
    if (layer == 0)      v = e0i[s] + e0h[s];
    else if (layer == 1) v = e1i[s] + e1h[s];
    else if (layer == 2) v = d0i[s] + d0h[s];
    else                 v = d1i[s] + d1h[s];
    biasp[i] = v;
  }
  if (i < 2048 * 7) {
    const int drow = i / 7, k = i % 7;
    const int wg = drow >> 6, nl = drow & 63, gate = nl >> 4, jj = nl & 15;
    const int s = gate * 512 + wg * 16 + jj;
    dwih0p[i] = dwih0[s * 7 + k];
  }
}

// ================= persistent recurrent kernel (software-pipelined across barriers) =================
__global__ __launch_bounds__(NTPB) void lstm_persist(
    const __bf16* __restrict__ emb, const __bf16* __restrict__ packW,
    const float* __restrict__ biasp, const float* __restrict__ dwih0p,
    const float* __restrict__ x_dec, const float* __restrict__ proj_w,
    const float* __restrict__ proj_b, __bf16* hbase, float* fb,
    float* out, unsigned* bar) {
  extern __shared__ char smem[];
  __bf16* W0 = (__bf16*)smem;
  __bf16* W1 = W0 + NSL * DMODEL;
  __bf16* W2 = W1 + NSL * DMODEL;
  __bf16* W3 = W2 + NSL * DMODEL;
  float* xs = (float*)(smem + SM_W);   // [448] dec Wih0 slice
  float* fbs = xs + 448;               // [896] fb stage

  const int tid = threadIdx.x, wg = blockIdx.x;
  const int lane = tid & 31, wave = tid >> 5;
  const int lq = lane & 15, half = lane >> 4;

  __bf16* h0buf[2] = {hbase, hbase + HBUF_ELEMS};
  __bf16* h1buf[2] = {hbase + 2 * HBUF_ELEMS, hbase + 3 * HBUF_ELEMS};

  // init: zero h ping buffer 0, fill decoder feedback seed
  {
    unsigned* p0 = (unsigned*)h0buf[0];
    unsigned* p1 = (unsigned*)h1buf[0];
    for (int i = tid; i < 1024; i += NTPB) { p0[wg * 1024 + i] = 0u; p1[wg * 1024 + i] = 0u; }
    if (wg == 0)
      for (int i = tid; i < BATCH * 7; i += NTPB)
        fb[i] = x_dec[((size_t)(i / 7) * DECT + DECROW) * 7 + (i % 7)];
  }

  // pin encoder weight slices in LDS for all 512 steps
  load_wlds(W0, packW + 0 * (size_t)MAT_ELEMS, wg, tid);  // enc Wih0
  load_wlds(W1, packW + 1 * (size_t)MAT_ELEMS, wg, tid);  // enc Whh0
  load_wlds(W2, packW + 2 * (size_t)MAT_ELEMS, wg, tid);  // enc Wih1
  load_wlds(W3, packW + 3 * (size_t)MAT_ELEMS, wg, tid);  // enc Whh1

  float br0[4], br1[4];
#pragma unroll
  for (int nt = 0; nt < 4; ++nt) {
    br0[nt] = biasp[0 * 2048 + wg * 64 + nt * 16 + lq];
    br1[nt] = biasp[1 * 2048 + wg * 64 + nt * 16 + lq];
  }

  float c0[8], c1[8];
#pragma unroll
  for (int v = 0; v < 8; ++v) { c0[v] = 0.f; c1[v] = 0.f; }

  unsigned phase = 0;
  gbar(bar, phase);     // init + zeros visible (syncthreads inside covers LDS loads)

  v8f acc[4], accN[4];

  // ---------------- encoder (1 GEMM on critical path, 3 hidden in barrier windows) ----------------
  // prologue: accN = bias0 + emb_0@Wih0 + h0(-1)@Whh0 (h0(-1)=0)
  acc_init(accN, br0);
  gemm_bf16(emb, (size_t)TSEQ * DMODEL, W0, accN, wave, lq, half);
  gemm_bf16(h0buf[0], DMODEL, W1, accN, wave, lq, half);

  for (int t = 0; t < TSEQ; ++t) {
    const int r = t & 1, w = 1 - r;
    // layer 0 update from pre-accumulated gates
    acc_copy(acc, accN);
    lstm_update(acc, c0, h0buf[w], wg, wave, lq, half);
    gsig(bar);
    // --- barrier-A window: independent GEMMs (h1(t-1) ready, emb constant) ---
    acc_init(acc, br1);
    gemm_bf16(h1buf[r], DMODEL, W3, acc, wave, lq, half);        // h1@Whh1
    if (t + 1 < TSEQ) {
      acc_init(accN, br0);
      gemm_bf16(emb + (size_t)(t + 1) * DMODEL, (size_t)TSEQ * DMODEL, W0,
                accN, wave, lq, half);                            // emb_{t+1}@Wih0
    }
    gwait(bar, phase);   // h0(t) now globally visible
    // critical-path GEMM
    gemm_bf16(h0buf[w], DMODEL, W2, acc, wave, lq, half);        // h0(t)@Wih1
    lstm_update(acc, c1, h1buf[w], wg, wave, lq, half);
    gsig(bar);
    // --- barrier-B window: h0(t)@Whh0 for step t+1 (h0(t) visible since waitA) ---
    if (t + 1 < TSEQ)
      gemm_bf16(h0buf[w], DMODEL, W1, accN, wave, lq, half);
    gwait(bar, phase);   // h1(t) now globally visible
  }

  // ---------------- swap to decoder weights ----------------
  load_wlds(W0, packW + 4 * (size_t)MAT_ELEMS, wg, tid);  // dec Whh0
  load_wlds(W1, packW + 5 * (size_t)MAT_ELEMS, wg, tid);  // dec Wih1
  load_wlds(W2, packW + 6 * (size_t)MAT_ELEMS, wg, tid);  // dec Whh1
  for (int i = tid; i < 448; i += NTPB) xs[i] = dwih0p[wg * 448 + i];
#pragma unroll
  for (int nt = 0; nt < 4; ++nt) {
    br0[nt] = biasp[2 * 2048 + wg * 64 + nt * 16 + lq];
    br1[nt] = biasp[3 * 2048 + wg * 64 + nt * 16 + lq];
  }
  __syncthreads();      // LDS weight swap visible to all waves of this WG

  // ---------------- decoder (recurrent GEMMs for td+1 hidden in barrier windows) ----------------
  v8f acc1N[4];
  // prologue: enc-final h0/h1 live in buffers[0] (visible since last enc barrier)
  acc_init(accN, br0);
  gemm_bf16(h0buf[0], DMODEL, W0, accN, wave, lq, half);   // h0@Whh0 for td=0
  acc_init(acc1N, br1);
  gemm_bf16(h1buf[0], DMODEL, W2, acc1N, wave, lq, half);  // h1@Whh1 for td=0

  for (int td = 0; td < PREDN; ++td) {
    const int r = td & 1, w = 1 - r;
    // layer 0: add x_prev @ Wih0^T (K=7, VALU) to pre-accumulated gates
    acc_copy(acc, accN);
    __syncthreads();
    for (int i = tid; i < BATCH * 7; i += NTPB) fbs[i] = fb[i];
    __syncthreads();
#pragma unroll
    for (int k = 0; k < 7; ++k) {
      float xv[8];
#pragma unroll
      for (int v = 0; v < 8; ++v) xv[v] = fbs[(wave * 16 + v + 8 * half) * 7 + k];
#pragma unroll
      for (int nt = 0; nt < 4; ++nt) {
        const float wk = xs[(nt * 16 + lq) * 7 + k];
#pragma unroll
        for (int v = 0; v < 8; ++v) acc[nt][v] += wk * xv[v];
      }
    }
    lstm_update(acc, c0, h0buf[w], wg, wave, lq, half);
    gbar(bar, phase);    // A: h0(td) visible
    // layer 1: critical-path GEMM on fresh h0
    acc_copy(acc, acc1N);
    gemm_bf16(h0buf[w], DMODEL, W1, acc, wave, lq, half);  // h0(td)@Wih1
    lstm_update(acc, c1, h1buf[w], wg, wave, lq, half);
    gsig(bar);
    // --- barrier-B window: h0(td)@Whh0 for td+1 ---
    if (td + 1 < PREDN) {
      acc_init(accN, br0);
      gemm_bf16(h0buf[w], DMODEL, W0, accN, wave, lq, half);
    }
    gwait(bar, phase);   // B: h1(td) visible
    do_proj(h1buf[w], proj_w, proj_b, out, fb, td, wg, tid);
    gsig(bar);
    // --- barrier-C window: h1(td)@Whh1 for td+1 ---
    if (td + 1 < PREDN) {
      acc_init(acc1N, br1);
      gemm_bf16(h1buf[w], DMODEL, W2, acc1N, wave, lq, half);
    }
    gwait(bar, phase);   // C: fb visible
  }
}

// ================= host entry =================
extern "C" void kernel_launch(void* const* d_in, const int* in_sizes, int n_in,
                              void* d_out, int out_size, void* d_ws, size_t ws_size,
                              hipStream_t stream) {
  (void)in_sizes; (void)n_in; (void)out_size; (void)ws_size;
  const float* x_enc   = (const float*)d_in[0];
  const float* x_mark  = (const float*)d_in[1];
  const float* x_dec   = (const float*)d_in[2];
  const float* conv_w  = (const float*)d_in[4];
  const float* time_w  = (const float*)d_in[5];
  const float* encWih0 = (const float*)d_in[6];
  const float* encWhh0 = (const float*)d_in[7];
  const float* encbih0 = (const float*)d_in[8];
  const float* encbhh0 = (const float*)d_in[9];
  const float* encWih1 = (const float*)d_in[10];
  const float* encWhh1 = (const float*)d_in[11];
  const float* encbih1 = (const float*)d_in[12];
  const float* encbhh1 = (const float*)d_in[13];
  const float* decWih0 = (const float*)d_in[14];
  const float* decWhh0 = (const float*)d_in[15];
  const float* decbih0 = (const float*)d_in[16];
  const float* decbhh0 = (const float*)d_in[17];
  const float* decWih1 = (const float*)d_in[18];
  const float* decWhh1 = (const float*)d_in[19];
  const float* decbih1 = (const float*)d_in[20];
  const float* decbhh1 = (const float*)d_in[21];
  const float* proj_w  = (const float*)d_in[22];
  const float* proj_b  = (const float*)d_in[23];

  char* ws = (char*)d_ws;
  unsigned* bar  = (unsigned*)(ws + WS_BAR);
  float* fb      = (float*)(ws + WS_FB);
  float* biasp   = (float*)(ws + WS_BIAS);
  float* dwih0p  = (float*)(ws + WS_DWIH0);
  __bf16* hbase  = (__bf16*)(ws + WS_H);
  __bf16* packW  = (__bf16*)(ws + WS_PACK);
  __bf16* emb    = (__bf16*)(ws + WS_EMB);

  hipMemsetAsync(bar, 0, 256, stream);  // deterministic barrier state per call

  embed_kernel<<<BATCH * TSEQ, NTPB, 0, stream>>>(x_enc, x_mark, conv_w, time_w, emb);

  const float* mats[7] = {encWih0, encWhh0, encWih1, encWhh1, decWhh0, decWih1, decWhh1};
  for (int m = 0; m < 7; ++m)
    pack_mat<<<2048, NTPB, 0, stream>>>(mats[m], packW + (size_t)m * MAT_ELEMS);

  pack_misc<<<56, NTPB, 0, stream>>>(encbih0, encbhh0, encbih1, encbhh1,
                                     decbih0, decbhh0, decbih1, decbhh1,
                                     decWih0, biasp, dwih0p);

  lstm_persist<<<NWG, NTPB, SMEM_TOTAL, stream>>>(emb, packW, biasp, dwih0p, x_dec,
                                                  proj_w, proj_b, hbase, fb,
                                                  (float*)d_out, bar);
}
